// CodebookModel_72481868087714
// MI455X (gfx1250) — matrix operand order repaired and checked by reading, not compiled, not run on hardware
//
#include <hip/hip_runtime.h>

// ---------------- problem constants ----------------
#define BDIM 16
#define SDIM 512
#define EDIM 768
#define NSUB 2
#define KCB  10000
#define DDIM 384
#define NROW (BDIM * SDIM)          // 8192 vectors
#define KT   128                    // codewords per K tile
#define KPAD (((KCB + KT - 1) / KT) * KT)   // 10112
#define NT   64                     // rows per workgroup
#define DC   64                     // D chunk staged in LDS
#define AP   72                     // LDS row pitch (bf16 elems): 144B = 9*16B, conflict-skewed

typedef __bf16 bf16_t;
typedef __attribute__((ext_vector_type(8)))  __bf16 v8bf;
typedef __attribute__((ext_vector_type(16))) __bf16 v16bf;
typedef __attribute__((ext_vector_type(8)))  float  v8f;
typedef __attribute__((ext_vector_type(4)))  int    v4i;

// gfx1250 async global->LDS copy (ASYNCcnt path). The builtin is declared in this
// toolchain; its first param is v4i in addrspace(1) (printed as "__device__"),
// second is the LDS destination in addrspace(3).
#if defined(__HIP_DEVICE_COMPILE__) && __has_builtin(__builtin_amdgcn_global_load_async_to_lds_b128)
#define HAVE_ASYNC_LDS 1
#else
#define HAVE_ASYNC_LDS 0
#endif

typedef __attribute__((address_space(1))) v4i gbl_v4i;
typedef __attribute__((address_space(3))) v4i lds_v4i;

__device__ inline void copy16_g2lds(const bf16_t* g, bf16_t* l) {
#if HAVE_ASYNC_LDS
  __builtin_amdgcn_global_load_async_to_lds_b128((gbl_v4i*)g, (lds_v4i*)l, 0, 0);
#else
  *(v8bf*)l = *(const v8bf*)g;
#endif
}

__device__ inline void wait_async_copies() {
#if HAVE_ASYNC_LDS
#if __has_builtin(__builtin_amdgcn_s_wait_asynccnt)
  __builtin_amdgcn_s_wait_asynccnt(0);
#else
  asm volatile("s_wait_asynccnt 0x0" ::: "memory");
#endif
#endif
}

// ---------------- helpers ----------------
__device__ inline v16bf frag_cat(v8bf a, v8bf b) {
  return __builtin_shufflevector(a, b, 0,1,2,3,4,5,6,7,8,9,10,11,12,13,14,15);
}
// A fragment: two 16B chunks at +0 and +16 elems (ISA 16-bit A 16x32 layout)
__device__ inline v16bf ld_frag_a(const bf16_t* p) {
  v8bf a = *(const v8bf*)p;
  v8bf b = *(const v8bf*)(p + 16);
  return frag_cat(a, b);
}
// B fragment: two 16B chunks at +0 and +16 elems (ISA 16-bit B 32x16 layout, row-major codeword)
__device__ inline v16bf ld_frag_b(const bf16_t* p) {
  v8bf a = *(const v8bf*)p;
  v8bf b = *(const v8bf*)(p + 16);
  return frag_cat(a, b);
}
__device__ inline v8f wmma_bf16(v16bf a, v16bf b, v8f c) {
  // (neg_a, A, neg_b, B, c_mod, C, reuse_a, reuse_b)
  return __builtin_amdgcn_wmma_f32_16x16x32_bf16(false, a, false, b, (short)0, c, false, false);
}

// ---------------- kernel 1: f32 embeddings -> bf16 hi/lo, layout [sub][n][d] ----------------
__global__ __launch_bounds__(256) void cvt_x_kernel(const float* __restrict__ emb,
                                                    bf16_t* __restrict__ Xh,
                                                    bf16_t* __restrict__ Xl) {
  size_t i = (size_t)blockIdx.x * 256 + threadIdx.x;       // over N*E
  if (i >= (size_t)NROW * EDIM) return;
  int n = (int)(i / EDIM);
  int r = (int)(i - (size_t)n * EDIM);
  int sub = r / DDIM;
  int d = r - sub * DDIM;
  float v = emb[i];
  bf16_t h = (bf16_t)v;
  bf16_t l = (bf16_t)(v - (float)h);
  size_t o = ((size_t)sub * NROW + n) * DDIM + d;
  Xh[o] = h;
  Xl[o] = l;
}

// ---------------- kernel 2: codebook -> bf16 hi/lo (padded to KPAD) + exact f32 c2 ----------------
__global__ __launch_bounds__(256) void cvt_c_kernel(const float* __restrict__ cb,
                                                    bf16_t* __restrict__ Ch,
                                                    bf16_t* __restrict__ Cl,
                                                    float* __restrict__ c2) {
  int row = blockIdx.x * 8 + (threadIdx.x >> 5);           // one wave per row
  int lane = threadIdx.x & 31;
  if (row >= NSUB * KPAD) return;
  int sub = row / KPAD;
  int k = row - sub * KPAD;
  size_t oo = (size_t)row * DDIM;
  float ss = 0.f;
  if (k < KCB) {
    const float* src = cb + ((size_t)sub * KCB + k) * DDIM;
    for (int d = lane; d < DDIM; d += 32) {
      float v = src[d];
      ss += v * v;
      bf16_t h = (bf16_t)v;
      Ch[oo + d] = h;
      Cl[oo + d] = (bf16_t)(v - (float)h);
    }
  } else {
    for (int d = lane; d < DDIM; d += 32) {
      Ch[oo + d] = (bf16_t)0.f;
      Cl[oo + d] = (bf16_t)0.f;
    }
  }
  #pragma unroll
  for (int m = 16; m >= 1; m >>= 1) ss += __shfl_xor(ss, m, 32);
  if (lane == 0) c2[row] = (k < KCB) ? ss : 3.0e38f;       // sentinel: pad rows never win argmin
}

// ---------------- kernel 3: WMMA score GEMM + fused argmin + gather ----------------
// 8 waves arranged 2(M) x 4(N): each wave owns a 32x32 sub-tile (2 m-tiles x 2 n-tiles),
// balancing A/B LDS fragment traffic at 16 ds_load_b128 per 12 WMMAs.
__global__ __launch_bounds__(256) void pq_argmin_kernel(const bf16_t* __restrict__ Xh,
                                                        const bf16_t* __restrict__ Xl,
                                                        const bf16_t* __restrict__ Ch,
                                                        const bf16_t* __restrict__ Cl,
                                                        const float*  __restrict__ c2g,
                                                        const float*  __restrict__ cbf32,
                                                        float* __restrict__ outq,
                                                        float* __restrict__ outidx) {
  __shared__ __align__(16) bf16_t sAh[NT][AP];
  __shared__ __align__(16) bf16_t sAl[NT][AP];
  __shared__ __align__(16) bf16_t sBh[KT][AP];
  __shared__ __align__(16) bf16_t sBl[KT][AP];
  __shared__ int sBest[NT];

  const int tid  = threadIdx.x;
  const int lane = tid & 31;
  const int wave = tid >> 5;
  const int wm   = wave & 1;        // which 32-row half of the 64-row tile
  const int wn   = wave >> 1;       // which 32-col group of the 128-col K tile
  const int n0   = blockIdx.x * NT;
  const int sub  = blockIdx.y;

  const int l15   = lane & 15;
  const int lhalf = lane >> 4;

  float bestS[2][8];
  int   bestK[2][8];
  #pragma unroll
  for (int mt = 0; mt < 2; ++mt)
    #pragma unroll
    for (int j = 0; j < 8; ++j) { bestS[mt][j] = __builtin_inff(); bestK[mt][j] = 0; }

  for (int kt = 0; kt < KPAD / KT; ++kt) {
    v8f acc[2][2] = {{v8f{}, v8f{}}, {v8f{}, v8f{}}};

    for (int dc = 0; dc < DDIM; dc += DC) {
      __syncthreads();
      // stage A chunk (64 x DC, hi+lo): 512 16B vectors, 2 per thread
      #pragma unroll
      for (int s = tid; s < NT * (DC / 8); s += 256) {
        int r = s >> 3, g = (s & 7) * 8;
        size_t go = ((size_t)sub * NROW + n0 + r) * DDIM + dc + g;
        copy16_g2lds(Xh + go, &sAh[r][g]);
        copy16_g2lds(Xl + go, &sAl[r][g]);
      }
      // stage B chunk (128 x DC, hi+lo): 1024 16B vectors, 4 per thread
      #pragma unroll
      for (int s = tid; s < KT * (DC / 8); s += 256) {
        int r = s >> 3, g = (s & 7) * 8;
        size_t go = ((size_t)sub * KPAD + (size_t)kt * KT + r) * DDIM + dc + g;
        copy16_g2lds(Ch + go, &sBh[r][g]);
        copy16_g2lds(Cl + go, &sBl[r][g]);
      }
      wait_async_copies();
      __syncthreads();

      #pragma unroll
      for (int kk = 0; kk < DC; kk += 32) {
        const int akb = (lhalf << 3) + kk;     // A lane k-offset: {0,8}+kk, chunks at +0/+16
        const int bkb = (lhalf << 3) + kk;     // B lane k-offset: {0,8}+kk, chunks at +0/+16
        v16bf ah[2], al[2];
        #pragma unroll
        for (int mt = 0; mt < 2; ++mt) {
          const int arow = wm * 32 + mt * 16 + l15;
          ah[mt] = ld_frag_a(&sAh[arow][akb]);
          al[mt] = ld_frag_a(&sAl[arow][akb]);
        }
        #pragma unroll
        for (int nt = 0; nt < 2; ++nt) {
          const int brow = wn * 32 + nt * 16 + l15;
          v16bf bh = ld_frag_b(&sBh[brow][bkb]);
          v16bf bl = ld_frag_b(&sBl[brow][bkb]);
          #pragma unroll
          for (int mt = 0; mt < 2; ++mt) {
            acc[mt][nt] = wmma_bf16(ah[mt], bh, acc[mt][nt]);   // hi*hi
            acc[mt][nt] = wmma_bf16(ah[mt], bl, acc[mt][nt]);   // hi*lo
            acc[mt][nt] = wmma_bf16(al[mt], bh, acc[mt][nt]);   // lo*hi
          }
        }
      }
    }

    // epilogue: score = c2[k] - 2*xc ; running argmin (registers + cached c2 loads)
    const int kbase = kt * KT + wn * 32;
    #pragma unroll
    for (int nt = 0; nt < 2; ++nt) {
      const int kg = kbase + nt * 16 + l15;
      const float c2v = c2g[(size_t)sub * KPAD + kg];
      #pragma unroll
      for (int mt = 0; mt < 2; ++mt) {
        #pragma unroll
        for (int j = 0; j < 8; ++j) {
          float sc = c2v - 2.0f * acc[mt][nt][j];
          if (sc < bestS[mt][j] || (sc == bestS[mt][j] && kg < bestK[mt][j])) {
            bestS[mt][j] = sc; bestK[mt][j] = kg;
          }
        }
      }
    }
  }

  // cross-thread argmin: 64 candidates per row, staged over the dead B tiles
  __syncthreads();
  float* redS = (float*)&sBh[0][0];   // [NT][64] floats = 16KB (fits in sBh)
  int*   redI = (int*)  &sBl[0][0];   // [NT][64] ints   = 16KB (fits in sBl)
  const int slot = wn * 16 + l15;     // 0..63
  #pragma unroll
  for (int mt = 0; mt < 2; ++mt) {
    #pragma unroll
    for (int j = 0; j < 8; ++j) {
      int r = wm * 32 + mt * 16 + (lhalf << 3) + j;  // C-layout: vgpr j -> M = j + 8*lanehalf
      redS[r * 64 + slot] = bestS[mt][j];
      redI[r * 64 + slot] = bestK[mt][j];
    }
  }
  __syncthreads();
  if (tid < NT) {
    float bs = redS[tid * 64];
    int   bk = redI[tid * 64];
    for (int s2 = 1; s2 < 64; ++s2) {
      float v = redS[tid * 64 + s2];
      int   k2 = redI[tid * 64 + s2];
      if (v < bs || (v == bs && k2 < bk)) { bs = v; bk = k2; }
    }
    sBest[tid] = bk;
    // indices output: (B,S,NSUB) row-major, written as float values (d_out is float-typed)
    outidx[(size_t)(n0 + tid) * NSUB + sub] = (float)bk;
  }
  __syncthreads();
  // gather winning original-f32 codebook rows: quantized[n, sub*384 + d]
  for (int i = tid; i < NT * DDIM; i += 256) {
    int r  = i / DDIM;
    int dd = i - r * DDIM;
    int bk = sBest[r];
    outq[(size_t)(n0 + r) * EDIM + sub * DDIM + dd] =
        cbf32[((size_t)sub * KCB + bk) * DDIM + dd];
  }
}

// ---------------- host launcher ----------------
extern "C" void kernel_launch(void* const* d_in, const int* in_sizes, int n_in,
                              void* d_out, int out_size, void* d_ws, size_t ws_size,
                              hipStream_t stream) {
  const float* emb = (const float*)d_in[0];   // (B,S,E) f32
  const float* cb  = (const float*)d_in[1];   // (NSUB,K,D) f32
  float* outq   = (float*)d_out;                          // B*S*E quantized
  float* outidx = (float*)d_out + (size_t)NROW * EDIM;    // B*S*NSUB indices (as float)

  // workspace carve (~57 MB): Xh,Xl (12.6MB ea), Ch,Cl (15.5MB ea), c2 (80KB)
  uintptr_t p = ((uintptr_t)d_ws + 255) & ~(uintptr_t)255;
  bf16_t* Xh = (bf16_t*)p; p += ((size_t)NSUB * NROW * DDIM * 2 + 255) & ~(size_t)255;
  bf16_t* Xl = (bf16_t*)p; p += ((size_t)NSUB * NROW * DDIM * 2 + 255) & ~(size_t)255;
  bf16_t* Ch = (bf16_t*)p; p += ((size_t)NSUB * KPAD * DDIM * 2 + 255) & ~(size_t)255;
  bf16_t* Cl = (bf16_t*)p; p += ((size_t)NSUB * KPAD * DDIM * 2 + 255) & ~(size_t)255;
  float*  c2 = (float*)p;

  // 1) split embeddings into bf16 hi/lo
  {
    size_t total = (size_t)NROW * EDIM;
    dim3 g((unsigned)((total + 255) / 256));
    hipLaunchKernelGGL(cvt_x_kernel, g, dim3(256), 0, stream, emb, Xh, Xl);
  }
  // 2) split codebooks + exact f32 c2 (padded rows -> sentinel)
  {
    dim3 g((unsigned)((NSUB * KPAD + 7) / 8));
    hipLaunchKernelGGL(cvt_c_kernel, g, dim3(256), 0, stream, cb, Ch, Cl, c2);
  }
  // 3) WMMA score GEMM + fused argmin + gather
  {
    dim3 g(NROW / NT, NSUB);
    hipLaunchKernelGGL(pq_argmin_kernel, g, dim3(256), 0, stream,
                       Xh, Xl, Ch, Cl, c2, cb, outq, outidx);
  }
}
